// Sequence_Modeling_40381282517740
// MI455X (gfx1250) — matrix operand order repaired and checked by reading, not compiled, not run on hardware
//
#include <hip/hip_runtime.h>
#include <cmath>

// ---------------------------------------------------------------------------
// Seq2seq (tanh-RNN encoder + attention decoder) for MI455X / gfx1250.
// All GEMMs use V_WMMA_F32_16X16X32_F16 (f16 A/B, f32 accumulate).
// Encoder recurrence runs as a single persistent workgroup: h in LDS,
// Whh pinned in VGPRs as preloaded WMMA B-fragments (read from L2 once).
// Decoder attention fully vectorized to b128 loads of f16.
// ---------------------------------------------------------------------------

#define DEV __device__ __forceinline__

typedef _Float16 h16;
typedef __attribute__((ext_vector_type(16))) _Float16 v16h;
typedef __attribute__((ext_vector_type(8)))  _Float16 v8h;
typedef __attribute__((ext_vector_type(8)))  float    v8f;

static constexpr int E    = 256;   // embedding dim
static constexpr int H    = 512;   // hidden dim
static constexpr int O    = 1024;  // output dim
static constexpr int B    = 32;    // batch
static constexpr int SE   = 512;   // encoder seq len
static constexpr int SD   = 128;   // decoder seq len
static constexpr int KIH  = H + E;   // 768  = dec_Wih K (cols: [ctx | e_t])
static constexpr int KCAT = KIH + H; // 1280 = [ctx | e_t | s] fused K
static constexpr int HP   = H + 8;   // LDS pad: stride 1040B -> bank spread

// ---- WMMA fragment loaders (contiguous k = (lane>>4)*16 + j in both A and B) ------

// A fragment: 16x32 f16 tile from row-major A (lda in elements, mult of 8).
// lane m = lane&15; halves j: K = (lane>>4)*16 + j.
DEV v16h load_a(const h16* A, int lda, int lane) {
  const int m = lane & 15, hf = (lane >> 4) & 1;
  const h16* base = A + (size_t)m * lda + hf * 16;
  v8h lo = *(const v8h*)(base);        // K = hf*16 + 0..7
  v8h hi = *(const v8h*)(base + 8);    // K = hf*16 + 8..15
  v16h r;
#pragma unroll
  for (int i = 0; i < 8; ++i) { r[i] = lo[i]; r[i + 8] = hi[i]; }
  return r;
}

// B fragment: 32x16 f16 tile where B[k][n] = W[n][k], W row-major (N x K).
// lane n = lane&15; halves j: K = (lane>>4)*16 + j  -> one contiguous v16h.
DEV v16h load_b(const h16* Wt, int ldw, int lane) {
  const int n = lane & 15, hf = (lane >> 4) & 1;
  return *(const v16h*)(Wt + (size_t)n * ldw + hf * 16);
}

DEV v8f wmma_f16(v16h a, v16h b, v8f c) {
  return __builtin_amdgcn_wmma_f32_16x16x32_f16(false, a, false, b,
                                                (short)0, c, false, false);
}

// Full K-loop tile: caller pre-offsets A by m0*lda and W by n0*ldw.
DEV v8f gemm_tile(const h16* A, int lda, const h16* Wt, int ldw, int K, int lane) {
  v8f c = {};
  for (int k0 = 0; k0 < K; k0 += 32)
    c = wmma_f16(load_a(A + k0, lda, lane), load_b(Wt + k0, ldw, lane), c);
  return c;
}

// ---- setup kernels --------------------------------------------------------

__global__ void k_f32_to_f16(const float* __restrict__ s, h16* __restrict__ d, int n) {
  int i = blockIdx.x * blockDim.x + threadIdx.x;
  if (i < n) d[i] = (h16)s[i];
}

__global__ void k_gather(const int* __restrict__ idx, const float* __restrict__ emb,
                         h16* __restrict__ dst, int rows) {
  int i = blockIdx.x * blockDim.x + threadIdx.x;
  if (i < rows * E) {
    int r = i / E, e = i - r * E;
    dst[i] = (h16)emb[(size_t)idx[r] * E + e];
  }
}

// W2[n, 0:768] = dec_Wih[n, :], W2[n, 768:1280] = dec_Whh[n, :]
__global__ void k_build_w2(const float* __restrict__ Wih, const float* __restrict__ Whh,
                           h16* __restrict__ W2) {
  int i = blockIdx.x * blockDim.x + threadIdx.x;
  if (i < H * KCAT) {
    int r = i / KCAT, c = i - r * KCAT;
    W2[i] = (h16)((c < KIH) ? Wih[(size_t)r * KIH + c]
                            : Whh[(size_t)r * H + (c - KIH)]);
  }
}

// ---- xW = enc_emb @ enc_Wih^T + bih : (16384 x 512), K=256 ----------------
// 4 M-tiles per wave sharing each B fragment: 256 M-groups x 32 N-tiles =
// 8192 waves -> 1024 blocks of 8 waves.
__global__ void __launch_bounds__(256)
k_gemm_xw(const h16* __restrict__ A, const h16* __restrict__ W,
          const float* __restrict__ bias, float* __restrict__ xW) {
  const int lane = threadIdx.x & 31;
  const int grp = blockIdx.x * 8 + (threadIdx.x >> 5); // 0..8191
  const int mg = grp >> 5, nt = grp & 31;
  const size_t m0 = (size_t)mg * 64;
  const int n0 = nt * 16;
  v8f c0 = {}, c1 = {}, c2 = {}, c3 = {};
#pragma unroll
  for (int k0 = 0; k0 < E; k0 += 32) {
    v16h bf = load_b(W + (size_t)n0 * E + k0, E, lane);
    c0 = wmma_f16(load_a(A + (m0 +  0) * E + k0, E, lane), bf, c0);
    c1 = wmma_f16(load_a(A + (m0 + 16) * E + k0, E, lane), bf, c1);
    c2 = wmma_f16(load_a(A + (m0 + 32) * E + k0, E, lane), bf, c2);
    c3 = wmma_f16(load_a(A + (m0 + 48) * E + k0, E, lane), bf, c3);
  }
  const int nn = lane & 15, hf = (lane >> 4) & 1, col = n0 + nn;
  const float bb = bias[col];
#pragma unroll
  for (int r = 0; r < 8; ++r) {
    const size_t m = m0 + r + hf * 8;
    xW[(m +  0) * H + col] = c0[r] + bb;
    xW[(m + 16) * H + col] = c1[r] + bb;
    xW[(m + 32) * H + col] = c2[r] + bb;
    xW[(m + 48) * H + col] = c3[r] + bb;
  }
}

// ---- persistent encoder: 512 steps, one workgroup of 32 waves -------------
// Wave w owns output columns [16w, 16w+16): holds its slice of Whh (16KB) in
// VGPRs as 16 preloaded B-fragments; state h lives in padded LDS.
// Also accumulates per-element running sum of h for the mlp1-mean init.
__global__ void __launch_bounds__(1024)
k_encoder(const float* __restrict__ xW, const h16* __restrict__ Whh,
          const float* __restrict__ bhh, h16* __restrict__ hid,
          h16* __restrict__ hbar) {
  __shared__ h16 hsh[B * HP];
  const int tid = threadIdx.x, lane = tid & 31, w = tid >> 5; // w = 0..31
  const int n0 = w * 16;
  const int nn = lane & 15, hf = (lane >> 4) & 1;
  const int col = n0 + nn;
  const float bh = bhh[col];

  // Pin this wave's Whh columns in registers: 16 v16h = 128 VGPRs.
  v16h bfr[16];
#pragma unroll
  for (int kk = 0; kk < 16; ++kk)
    bfr[kk] = load_b(Whh + (size_t)n0 * H + kk * 32, H, lane);

  for (int i = tid; i < B * HP; i += 1024) hsh[i] = (h16)0.0f;
  __syncthreads();

  float acc0[8], acc1[8];
#pragma unroll
  for (int r = 0; r < 8; ++r) { acc0[r] = 0.0f; acc1[r] = 0.0f; }

  for (int t = 0; t < SE; ++t) {
    v8f c0 = {}, c1 = {};
#pragma unroll
    for (int kk = 0; kk < 16; ++kk) {
      v16h a0 = load_a(hsh + kk * 32, HP, lane);                      // rows 0..15
      v16h a1 = load_a(hsh + (size_t)16 * HP + kk * 32, HP, lane);    // rows 16..31
      c0 = wmma_f16(a0, bfr[kk], c0);
      c1 = wmma_f16(a1, bfr[kk], c1);
    }
    __syncthreads();  // everyone done reading h before overwrite
#pragma unroll
    for (int r = 0; r < 8; ++r) {
      const int m0 = r + hf * 8, m1 = m0 + 16;
      float v0 = tanhf(c0[r] + xW[((size_t)m0 * SE + t) * H + col] + bh);
      float v1 = tanhf(c1[r] + xW[((size_t)m1 * SE + t) * H + col] + bh);
      hsh[m0 * HP + col] = (h16)v0;
      hsh[m1 * HP + col] = (h16)v1;
      hid[((size_t)m0 * SE + t) * H + col] = (h16)v0;
      hid[((size_t)m1 * SE + t) * H + col] = (h16)v1;
      acc0[r] += v0; acc1[r] += v1;
    }
    __syncthreads();
  }
  const float inv = 1.0f / (float)SE;
#pragma unroll
  for (int r = 0; r < 8; ++r) {
    hbar[(r + hf * 8) * H + col]      = (h16)(acc0[r] * inv);
    hbar[(r + hf * 8 + 16) * H + col] = (h16)(acc1[r] * inv);
  }
}

// ---- s0 = hbar @ mlp1_W^T + mlp1_b : 32x512, K=512 ------------------------
__global__ void __launch_bounds__(256)
k_gemm_state(const h16* __restrict__ A, const h16* __restrict__ W,
             const float* __restrict__ bias, h16* __restrict__ hstate) {
  const int lane = threadIdx.x & 31;
  const int w = blockIdx.x * 8 + (threadIdx.x >> 5); // 0..63
  const int mt = w >> 5, nt = w & 31;
  const int m0 = mt * 16, n0 = nt * 16;
  v8f c = gemm_tile(A + (size_t)m0 * H, H, W + (size_t)n0 * H, H, H, lane);
  const int nn = lane & 15, hf = (lane >> 4) & 1, col = n0 + nn;
  const float bb = bias[col];
#pragma unroll
  for (int r = 0; r < 8; ++r)
    hstate[(m0 + r + hf * 8) * H + col] = (h16)(c[r] + bb);
}

// ---- decoder attention: one block per batch row ---------------------------
// Builds inp2[b] = [ctx(512) | e_t(256) | s(512)] in f16.
// Score pass: thread tid owns row tid, v8h (b128) loads.
// Ctx pass: thread (seg,chunk) owns 8 columns x 64 rows, v8h loads,
// partials reduced through LDS.
__global__ void __launch_bounds__(512)
k_attn(const h16* __restrict__ hid, const h16* __restrict__ hstate,
       const h16* __restrict__ decEmb, h16* __restrict__ inp2, int t) {
  __shared__ float sv[H];
  __shared__ float red[SE];
  __shared__ float alpha[SE];
  __shared__ float cbuf[SE * 8];   // 16 KB: partial ctx sums [thread][8 cols]
  const int b = blockIdx.x, tid = threadIdx.x;

  sv[tid] = (float)hstate[b * H + tid];
  inp2[b * KCAT + KIH + tid] = hstate[b * H + tid];                    // s slot
  if (tid < E)
    inp2[b * KCAT + H + tid] = decEmb[((size_t)b * SD + t) * E + tid]; // e_t
  __syncthreads();

  // scores: row tid of enc_hidden[b] dot s
  const h16* row = hid + ((size_t)b * SE + tid) * H;
  float acc = 0.0f;
#pragma unroll 4
  for (int h8 = 0; h8 < H; h8 += 8) {
    v8h v = *(const v8h*)(row + h8);
#pragma unroll
    for (int i = 0; i < 8; ++i) acc += (float)v[i] * sv[h8 + i];
  }
  red[tid] = acc;
  __syncthreads();
  for (int s = 256; s > 0; s >>= 1) {
    if (tid < s) red[tid] = fmaxf(red[tid], red[tid + s]);
    __syncthreads();
  }
  const float mx = red[0];
  __syncthreads();
  const float e = __expf(acc - mx);
  red[tid] = e;
  __syncthreads();
  for (int s = 256; s > 0; s >>= 1) {
    if (tid < s) red[tid] += red[tid + s];
    __syncthreads();
  }
  alpha[tid] = e / red[0];
  __syncthreads();

  // ctx: seg = 8-column slab, chunk = 64-row slab of alpha-weighted rows
  const int seg = tid >> 3;         // 0..63 -> columns [seg*8, seg*8+8)
  const int chunk = tid & 7;        // 0..7  -> rows [chunk*64, chunk*64+64)
  float pc[8];
#pragma unroll
  for (int i = 0; i < 8; ++i) pc[i] = 0.0f;
  const h16* base = hid + (size_t)b * SE * H + seg * 8;
#pragma unroll 4
  for (int si = chunk * 64; si < chunk * 64 + 64; ++si) {
    const float al = alpha[si];
    v8h v = *(const v8h*)(base + (size_t)si * H);
#pragma unroll
    for (int i = 0; i < 8; ++i) pc[i] += al * (float)v[i];
  }
#pragma unroll
  for (int i = 0; i < 8; ++i) cbuf[tid * 8 + i] = pc[i];
  __syncthreads();

  // final: thread tid reduces 8 chunk-partials for column tid
  {
    const int cseg = tid >> 3, ci = tid & 7;
    float c = 0.0f;
#pragma unroll
    for (int k = 0; k < 8; ++k) c += cbuf[(cseg * 8 + k) * 8 + ci];
    inp2[b * KCAT + tid] = (h16)c;
  }
}

// ---- h = tanh(inp2 @ [Wih|Whh]^T + bih + bhh) : 32x512, K=1280 ------------
__global__ void __launch_bounds__(256)
k_dec_rnn(const h16* __restrict__ inp2, const h16* __restrict__ W2,
          const float* __restrict__ bih, const float* __restrict__ bhh,
          h16* __restrict__ hstate) {
  const int lane = threadIdx.x & 31;
  const int w = blockIdx.x * 8 + (threadIdx.x >> 5); // 0..63
  const int mt = w >> 5, nt = w & 31;
  const int m0 = mt * 16, n0 = nt * 16;
  v8f c = gemm_tile(inp2 + (size_t)m0 * KCAT, KCAT,
                    W2 + (size_t)n0 * KCAT, KCAT, KCAT, lane);
  const int nn = lane & 15, hf = (lane >> 4) & 1, col = n0 + nn;
  const float bb = bih[col] + bhh[col];
#pragma unroll
  for (int r = 0; r < 8; ++r)
    hstate[(m0 + r + hf * 8) * H + col] = (h16)tanhf(c[r] + bb);
}

// ---- pred = h @ mlp2_W^T + mlp2_b : 32x1024, K=512 ------------------------
__global__ void __launch_bounds__(256)
k_pred(const h16* __restrict__ hstate, const h16* __restrict__ W,
       const float* __restrict__ bias, float* __restrict__ out, int t) {
  const int lane = threadIdx.x & 31;
  const int w = blockIdx.x * 8 + (threadIdx.x >> 5); // 0..127
  const int mt = w >> 6, nt = w & 63;
  const int m0 = mt * 16, n0 = nt * 16;
  v8f c = gemm_tile(hstate + (size_t)m0 * H, H, W + (size_t)n0 * H, H, H, lane);
  const int nn = lane & 15, hf = (lane >> 4) & 1, col = n0 + nn;
  const float bb = bias[col];
#pragma unroll
  for (int r = 0; r < 8; ++r)
    out[((size_t)(m0 + r + hf * 8) * SD + t) * O + col] = c[r] + bb;
}

// ---------------------------------------------------------------------------

extern "C" void kernel_launch(void* const* d_in, const int* in_sizes, int n_in,
                              void* d_out, int out_size, void* d_ws, size_t ws_size,
                              hipStream_t stream) {
  (void)in_sizes; (void)n_in; (void)out_size; (void)ws_size;
  const int*   enc_x = (const int*)  d_in[0];
  const int*   dec_y = (const int*)  d_in[1];
  const float* emb   = (const float*)d_in[2];
  const float* eWih  = (const float*)d_in[3];
  const float* eWhh  = (const float*)d_in[4];
  const float* ebih  = (const float*)d_in[5];
  const float* ebhh  = (const float*)d_in[6];
  const float* m1W   = (const float*)d_in[7];
  const float* m1b   = (const float*)d_in[8];
  const float* dWih  = (const float*)d_in[9];
  const float* dWhh  = (const float*)d_in[10];
  const float* dbih  = (const float*)d_in[11];
  const float* dbhh  = (const float*)d_in[12];
  const float* m2W   = (const float*)d_in[13];
  const float* m2b   = (const float*)d_in[14];
  float* out = (float*)d_out;

  // workspace carve-out (256B aligned); total ~65 MB, L2-resident
  char* ws = (char*)d_ws;
  size_t off = 0;
  auto take = [&](size_t bytes) -> char* {
    char* p = ws + off;
    off = (off + bytes + 255) & ~(size_t)255;
    return p;
  };
  h16*   encEmb = (h16*)take((size_t)B * SE * E * sizeof(h16));
  h16*   decEmb = (h16*)take((size_t)B * SD * E * sizeof(h16));
  h16*   WihH   = (h16*)take((size_t)H * E * sizeof(h16));
  h16*   WhhH   = (h16*)take((size_t)H * H * sizeof(h16));
  h16*   m1H    = (h16*)take((size_t)H * H * sizeof(h16));
  h16*   m2H    = (h16*)take((size_t)O * H * sizeof(h16));
  h16*   W2H    = (h16*)take((size_t)H * KCAT * sizeof(h16));
  float* xW     = (float*)take((size_t)B * SE * H * sizeof(float));
  h16*   hidH   = (h16*)take((size_t)B * SE * H * sizeof(h16));
  h16*   hbarH  = (h16*)take((size_t)B * H * sizeof(h16));
  h16*   hstate = (h16*)take((size_t)B * H * sizeof(h16));
  h16*   inp2   = (h16*)take((size_t)B * KCAT * sizeof(h16));

  auto blk = [](int n) { return (n + 255) / 256; };

  // weight conversions + embedding gathers
  k_f32_to_f16<<<blk(H * E), 256, 0, stream>>>(eWih, WihH, H * E);
  k_f32_to_f16<<<blk(H * H), 256, 0, stream>>>(eWhh, WhhH, H * H);
  k_f32_to_f16<<<blk(H * H), 256, 0, stream>>>(m1W, m1H, H * H);
  k_f32_to_f16<<<blk(O * H), 256, 0, stream>>>(m2W, m2H, O * H);
  k_build_w2<<<blk(H * KCAT), 256, 0, stream>>>(dWih, dWhh, W2H);
  k_gather<<<blk(B * SE * E), 256, 0, stream>>>(enc_x, emb, encEmb, B * SE);
  k_gather<<<blk(B * SD * E), 256, 0, stream>>>(dec_y, emb, decEmb, B * SD);

  // xW = enc_emb @ Wih^T + bih  (8192 waves, 4 M-tiles each)
  k_gemm_xw<<<1024, 256, 0, stream>>>(encEmb, WihH, ebih, xW);

  // persistent encoder recurrence (one WG, 512 steps internally)
  k_encoder<<<1, 1024, 0, stream>>>(xW, WhhH, ebhh, hidH, hbarH);

  // decoder init state s0 = hbar @ mlp1_W^T + mlp1_b
  k_gemm_state<<<8, 256, 0, stream>>>(hbarH, m1H, m1b, hstate);

  // decoder loop: attention -> fused RNN GEMM -> prediction GEMM
  for (int t = 0; t < SD; ++t) {
    k_attn<<<B, 512, 0, stream>>>(hidH, hstate, decEmb, inp2, t);
    k_dec_rnn<<<8, 256, 0, stream>>>(inp2, W2H, dbih, dbhh, hstate);
    k_pred<<<16, 256, 0, stream>>>(hstate, m2H, m2b, out, t);
  }
}
